// MResConv_49383533969434
// MI455X (gfx1250) — matrix operand order repaired and checked by reading, not compiled, not run on hardware
//
#include <hip/hip_runtime.h>
#include <hip/hip_bf16.h>

// Problem constants (match reference)
#define B_N     2
#define E_N     120000
#define CIN_N   64
#define COUT_N  128
#define EPS_BN  1e-5f

typedef __attribute__((ext_vector_type(16))) __bf16 v16bf;
typedef __attribute__((ext_vector_type(8)))  float  v8f;

// ---------- bf16 helpers (RNE convert via bit ops; no reliance on hw cvt) ----
__device__ __forceinline__ __bf16 f2bf(float f) {
  unsigned u = __builtin_bit_cast(unsigned, f);
  unsigned r = u + 0x7FFFu + ((u >> 16) & 1u);
  unsigned short h = (unsigned short)(r >> 16);
  return __builtin_bit_cast(__bf16, h);
}
__device__ __forceinline__ float bf2f(__bf16 h) {
  unsigned short us = __builtin_bit_cast(unsigned short, h);
  return __builtin_bit_cast(float, (unsigned)us << 16);
}

// ---------- WMMA 16-bit fragment index maps (per CDNA5 ISA 7.12.2) ----------
// A-matrix (16x32 MxK): lane L holds row M=L%16; elem i -> K = i + 8*half + 8*(i>=8)
__device__ __forceinline__ int ka_of(int i, int half) {
  return i + 8 * half + ((i >= 8) ? 8 : 0);
}
// B-matrix (32x16 KxN): lane L holds col N=L%16; elem i -> K = i + 8*half + 8*(i>=8)
// Store one feature value f[k][e] into LDS at its B-fragment slot.
__device__ __forceinline__ void stB(__bf16* ldsB, int k, int el, float v) {
  int kk = k >> 5, kl = k & 31;
  int half = (kl >> 3) & 1, i = (kl & 7) + ((kl >> 4) << 3);
  ldsB[(size_t)((kk << 5) + el + (half << 4)) * 16 + i] = f2bf(v);
}

// ---------- Kernel 1: pack W (f32 [COUT, K]) -> bf16 A-fragment layout -------
// Wp layout: [m_tile][k_chunk][lane(32)][elem(16)] bf16, so each lane reads its
// whole fragment as 32 contiguous bytes (2x global_load_b128).
template <int K>
__global__ __launch_bounds__(256) void pack_w(const float* __restrict__ W,
                                              __bf16* __restrict__ Wp,
                                              int total) {
  int idx = blockIdx.x * 256 + threadIdx.x;
  if (idx >= total) return;
  const int KCH = K / 32;
  int i    = idx & 15;
  int L    = (idx >> 4) & 31;
  int frag = idx >> 9;
  int kk   = frag % KCH;
  int mt   = frag / KCH;
  int half = L >> 4;
  int o    = mt * 16 + (L & 15);
  int k    = kk * 32 + ka_of(i, half);
  Wp[idx] = f2bf(W[(size_t)o * K + k]);
}

// ---------- Kernel 2: conv0 via bf16 WMMA, f32 accumulate --------------------
__global__ __launch_bounds__(256) void conv0_wmma(const float* __restrict__ x,
                                                  const int* __restrict__ eidx,
                                                  const __bf16* __restrict__ Wp,
                                                  float* __restrict__ y0) {
  const int KCH = (CIN_N * 5) / 32;                 // 10
  __shared__ __align__(32) __bf16 ldsB[KCH * 32 * 16];  // 10 KB
  __shared__ int snbr[16][4];

  const int tid = threadIdx.x;
  const int b   = blockIdx.y;
  const int e0  = blockIdx.x * 16;

  if (tid < 64) {
    int el = tid >> 2, j = tid & 3;
    snbr[el][j] = eidx[((size_t)b * E_N + e0 + el) * 4 + j];
  }
  __syncthreads();

  // Build B (features) in LDS: 64 channels x 16 edges, 5 features each
  for (int p = tid; p < CIN_N * 16; p += 256) {
    int c = p >> 4, el = p & 15;
    const float* xc = x + ((size_t)b * CIN_N + c) * E_N;
    float xs = xc[e0 + el];
    float a  = xc[snbr[el][0]];
    float bb = xc[snbr[el][1]];
    float cc = xc[snbr[el][2]];
    float dd = xc[snbr[el][3]];
    int kb = c * 5;
    stB(ldsB, kb + 0, el, xs);
    stB(ldsB, kb + 1, el, a + cc);
    stB(ldsB, kb + 2, el, bb + dd);
    stB(ldsB, kb + 3, el, fabsf(a - cc));
    stB(ldsB, kb + 4, el, fabsf(bb - dd));
  }
  __syncthreads();

  const int wave = tid >> 5, lane = tid & 31;
  v8f acc = {};
  const v16bf* Af = ((const v16bf*)Wp) + (size_t)(wave * KCH) * 32 + lane;
  const v16bf* Bf = ((const v16bf*)ldsB) + lane;
#pragma unroll
  for (int kk = 0; kk < KCH; ++kk) {
    v16bf afrag = Af[(size_t)kk * 32];
    v16bf bfrag = Bf[(size_t)kk * 32];
    acc = __builtin_amdgcn_wmma_f32_16x16x32_bf16(
        false, afrag, false, bfrag, (short)0, acc, false, false);
  }

  const int n = lane & 15, half = lane >> 4;
  float* yo = y0 + ((size_t)b * COUT_N + wave * 16 + 8 * half) * E_N + e0 + n;
#pragma unroll
  for (int j = 0; j < 8; ++j) yo[(size_t)j * E_N] = acc[j];
}

// ---------- Kernel 3a/3b: per-channel BN statistics of relu(y0) --------------
__global__ void zero_stats(float* __restrict__ stats) {
  stats[threadIdx.x] = 0.0f;  // 1 block x 256 threads covers 2*COUT floats
}

__global__ __launch_bounds__(256) void reduce_stats(const float* __restrict__ y0,
                                                    float* __restrict__ stats) {
  const int c     = blockIdx.x;
  const int slice = blockIdx.y;
  const int SL    = 8;
  const int per   = (B_N * E_N) / SL;  // 30000
  float s1 = 0.f, s2 = 0.f;
  for (int t = slice * per + threadIdx.x; t < (slice + 1) * per; t += 256) {
    int b = t / E_N, e = t % E_N;
    float v = y0[((size_t)b * COUT_N + c) * E_N + e];
    v = fmaxf(v, 0.f);
    s1 += v;
    s2 += v * v;
  }
  __shared__ float sh1[256], sh2[256];
  sh1[threadIdx.x] = s1;
  sh2[threadIdx.x] = s2;
  __syncthreads();
  for (int off = 128; off > 0; off >>= 1) {
    if (threadIdx.x < off) {
      sh1[threadIdx.x] += sh1[threadIdx.x + off];
      sh2[threadIdx.x] += sh2[threadIdx.x + off];
    }
    __syncthreads();
  }
  if (threadIdx.x == 0) {
    atomicAdd(&stats[c], sh1[0]);
    atomicAdd(&stats[COUT_N + c], sh2[0]);
  }
}

// ---------- Kernel 4: z = bf16( bn(relu(y0)) ) -------------------------------
__global__ __launch_bounds__(256) void bn_apply(const float* __restrict__ y0,
                                                const float* __restrict__ stats,
                                                const float* __restrict__ gamma,
                                                const float* __restrict__ beta,
                                                __bf16* __restrict__ z) {
  size_t idx = (size_t)blockIdx.x * 256 + threadIdx.x;
  const size_t total = (size_t)B_N * COUT_N * E_N;
  if (idx >= total) return;
  int c = (int)((idx / E_N) % COUT_N);
  const float invN = 1.0f / (float)(B_N * E_N);
  float m   = stats[c] * invN;
  float var = stats[COUT_N + c] * invN - m * m;
  float sc  = gamma[c] * rsqrtf(var + EPS_BN);
  float r   = fmaxf(y0[idx], 0.f);
  z[idx] = f2bf((r - m) * sc + beta[c]);
}

// ---------- Kernel 5: conv1 via WMMA + residual + relu -> out ----------------
__global__ __launch_bounds__(256) void conv1_wmma(const __bf16* __restrict__ z,
                                                  const int* __restrict__ eidx,
                                                  const __bf16* __restrict__ Wp,
                                                  const float* __restrict__ y0,
                                                  float* __restrict__ out) {
  const int KCH = (COUT_N * 5) / 32;                // 20
  __shared__ __align__(32) __bf16 ldsB[KCH * 32 * 16];  // 20 KB
  __shared__ int snbr[16][4];

  const int tid = threadIdx.x;
  const int b   = blockIdx.y;
  const int e0  = blockIdx.x * 16;

  if (tid < 64) {
    int el = tid >> 2, j = tid & 3;
    snbr[el][j] = eidx[((size_t)b * E_N + e0 + el) * 4 + j];
  }
  __syncthreads();

  for (int p = tid; p < COUT_N * 16; p += 256) {
    int c = p >> 4, el = p & 15;
    const __bf16* zc = z + ((size_t)b * COUT_N + c) * E_N;
    float xs = bf2f(zc[e0 + el]);
    float a  = bf2f(zc[snbr[el][0]]);
    float bb = bf2f(zc[snbr[el][1]]);
    float cc = bf2f(zc[snbr[el][2]]);
    float dd = bf2f(zc[snbr[el][3]]);
    int kb = c * 5;
    stB(ldsB, kb + 0, el, xs);
    stB(ldsB, kb + 1, el, a + cc);
    stB(ldsB, kb + 2, el, bb + dd);
    stB(ldsB, kb + 3, el, fabsf(a - cc));
    stB(ldsB, kb + 4, el, fabsf(bb - dd));
  }
  __syncthreads();

  const int wave = tid >> 5, lane = tid & 31;
  v8f acc = {};
  const v16bf* Af = ((const v16bf*)Wp) + (size_t)(wave * KCH) * 32 + lane;
  const v16bf* Bf = ((const v16bf*)ldsB) + lane;
#pragma unroll
  for (int kk = 0; kk < KCH; ++kk) {
    v16bf afrag = Af[(size_t)kk * 32];
    v16bf bfrag = Bf[(size_t)kk * 32];
    acc = __builtin_amdgcn_wmma_f32_16x16x32_bf16(
        false, afrag, false, bfrag, (short)0, acc, false, false);
  }

  const int n = lane & 15, half = lane >> 4;
  size_t base = ((size_t)b * COUT_N + wave * 16 + 8 * half) * E_N + e0 + n;
  const float* yr = y0 + base;
  float* oo = out + base;
#pragma unroll
  for (int j = 0; j < 8; ++j)
    oo[(size_t)j * E_N] = fmaxf(acc[j] + yr[(size_t)j * E_N], 0.f);
}

// ---------- Launch ----------------------------------------------------------
extern "C" void kernel_launch(void* const* d_in, const int* in_sizes, int n_in,
                              void* d_out, int out_size, void* d_ws, size_t ws_size,
                              hipStream_t stream) {
  const float* x     = (const float*)d_in[0];
  const int*   eidx  = (const int*)d_in[1];
  const float* W0    = (const float*)d_in[2];
  const float* W1    = (const float*)d_in[3];
  const float* gamma = (const float*)d_in[4];
  const float* beta  = (const float*)d_in[5];
  float* out = (float*)d_out;

  // Workspace carve-up (bytes):
  //   y0   : B*COUT*E f32      = 122,880,000
  //   z    : B*COUT*E bf16     =  61,440,000
  //   Wp0  : 8*10*512  bf16    =      81,920
  //   Wp1  : 8*20*512  bf16    =     163,840
  //   stats: 2*COUT    f32     =       1,024
  char* ws = (char*)d_ws;
  float*  y0    = (float*)(ws);
  __bf16* z     = (__bf16*)(ws + 122880000);
  __bf16* Wp0   = (__bf16*)(ws + 184320000);
  __bf16* Wp1   = (__bf16*)(ws + 184401920);
  float*  stats = (float*)(ws + 184565760);

  pack_w<320><<<160, 256, 0, stream>>>(W0, Wp0, 8 * 10 * 512);
  pack_w<640><<<320, 256, 0, stream>>>(W1, Wp1, 8 * 20 * 512);
  zero_stats<<<1, 256, 0, stream>>>(stats);

  dim3 cgrid(E_N / 16, B_N);
  conv0_wmma<<<cgrid, 256, 0, stream>>>(x, eidx, Wp0, y0);
  reduce_stats<<<dim3(COUT_N, 8), 256, 0, stream>>>(y0, stats);

  unsigned nb = (unsigned)(((size_t)B_N * COUT_N * E_N + 255) / 256);
  bn_apply<<<nb, 256, 0, stream>>>(y0, stats, gamma, beta, z);

  conv1_wmma<<<cgrid, 256, 0, stream>>>(z, eidx, Wp1, y0, out);
}
